// feature_extraction_70540542869680
// MI455X (gfx1250) — matrix operand order repaired and checked by reading, not compiled, not run on hardware
//
#include <hip/hip_runtime.h>

#define NPTS 2048
#define NB   16
#define Gc   24

typedef __attribute__((ext_vector_type(16))) _Float16 v16h;
typedef __attribute__((ext_vector_type(8)))  float    v8f;
typedef __attribute__((ext_vector_type(2)))  float    v2f;

static __host__ __device__ constexpr int rup32c(int x) { return (x + 31) & ~31; }

// Pair-interleaved LDS layout for activation matrices:
//   element (k, col) of a [K][LDC] matrix lives at (k>>1)*(2*LDC) + 2*col + (k&1)
// so the (k, k+1) half-pair a WMMA fragment VGPR needs is one aligned dword.
template <int LDC>
__device__ __forceinline__ int ipos(int k, int col) {
  return (k >> 1) * (2 * LDC) + 2 * col + (k & 1);
}

// ---------------------------------------------------------------- utilities
__global__ void conv1_scalar_kernel(const float* __restrict__ X,
                                    const float* __restrict__ W,
                                    const float* __restrict__ Bias,
                                    float* __restrict__ Y) {
  int t = blockIdx.x * blockDim.x + threadIdx.x;
  if (t >= NB * 24 * NPTS) return;
  int n = t & (NPTS - 1);
  int r = t >> 11;
  int o = r % 24, b = r / 24;
  float acc = Bias[o];
#pragma unroll
  for (int c = 0; c < 3; ++c)
    acc += W[o * 3 + c] * X[((size_t)b * 3 + c) * NPTS + n];
  Y[((size_t)b * 24 + o) * NPTS + n] = acc > 0.f ? acc : 0.f;
}

template <int C>
__global__ void sqnorm_kernel(const float* __restrict__ X,
                              float* __restrict__ SQ) {
  int t = blockIdx.x * blockDim.x + threadIdx.x;
  if (t >= NB * NPTS) return;
  int n = t & (NPTS - 1), b = t >> 11;
  const float* xp = X + (size_t)b * C * NPTS + n;
  float s = 0.f;
#pragma unroll
  for (int c = 0; c < C; ++c) { float v = xp[(size_t)c * NPTS]; s += v * v; }
  SQ[t] = s;
}

__global__ void copych_kernel(const float* __restrict__ S, int Cs,
                              float* __restrict__ D, int Ctot, int off) {
  int t = blockIdx.x * blockDim.x + threadIdx.x;
  if (t >= NB * Cs * NPTS) return;
  int n = t & (NPTS - 1);
  int r = t >> 11;
  int c = r % Cs, b = r / Cs;
  D[((size_t)b * Ctot + off + c) * NPTS + n] = S[t];
}

// ------------------------------------------------- KNN: f32 WMMA inner products
// One wave per block. Computes P[i,j] = x_i . x_j for a 16-row x 128-col tile.
template <int C>
__global__ void knn_dist_kernel(const float* __restrict__ X, int b,
                                float* __restrict__ P) {
  const int lane = threadIdx.x;
  const int m = lane & 15, h = lane >> 4;
  const int i0 = blockIdx.x * 16;
  const int j0base = blockIdx.y * 128;
  const float* Xb = X + (size_t)b * C * NPTS;

  constexpr int KST = C / 4;
  v2f afrag[KST];
#pragma unroll
  for (int kt = 0; kt < KST; ++kt) {
    int k = kt * 4 + 2 * h;            // lanes 0-15: K=k0,k0+1; 16-31: k0+2,k0+3
    afrag[kt][0] = Xb[(size_t)k * NPTS + i0 + m];
    afrag[kt][1] = Xb[(size_t)(k + 1) * NPTS + i0 + m];
  }
  for (int jt = 0; jt < 8; ++jt) {
    const int j0 = j0base + jt * 16;
    v8f acc = {};
#pragma unroll
    for (int kt = 0; kt < KST; ++kt) {
      int k = kt * 4 + 2 * h;
      v2f bfrag;
      bfrag[0] = Xb[(size_t)k * NPTS + j0 + m];
      bfrag[1] = Xb[(size_t)(k + 1) * NPTS + j0 + m];
      acc = __builtin_amdgcn_wmma_f32_16x16x4_f32(
          false, afrag[kt], false, bfrag, (short)0, acc, false, false);
    }
#pragma unroll
    for (int r = 0; r < 8; ++r) {
      int row = r + 8 * h;                       // C/D layout: VGPR r -> M = r+8h
      P[(size_t)(i0 + row) * NPTS + j0 + m] = acc[r];
    }
  }
}

// One wave per row: 16 nearest neighbors by key sq[j] - 2*P[i,j], self excluded.
__global__ void knn_topk_kernel(const float* __restrict__ P,
                                const float* __restrict__ SQ, int b,
                                int* __restrict__ IDX) {
  const int i = blockIdx.x;
  const int lane = threadIdx.x;
  const float* prow = P + (size_t)i * NPTS;
  const float* sq = SQ + (size_t)b * NPTS;

  float val[17]; int id[17];
#pragma unroll
  for (int t = 0; t < 17; ++t) { val[t] = 3.0e38f; id[t] = -1; }
  for (int j = lane; j < NPTS; j += 32) {
    if (j == i) continue;
    float key = sq[j] - 2.0f * prow[j];
    if (key < val[16]) {
      val[16] = key; id[16] = j;
#pragma unroll
      for (int t = 16; t > 0; --t) {
        if (val[t] < val[t - 1]) {
          float tv = val[t]; val[t] = val[t - 1]; val[t - 1] = tv;
          int ti = id[t]; id[t] = id[t - 1]; id[t - 1] = ti;
        }
      }
    }
  }
  __shared__ float lval[32 * 17];
  __shared__ int   lid [32 * 17];
#pragma unroll
  for (int t = 0; t < 17; ++t) { lval[lane * 17 + t] = val[t]; lid[lane * 17 + t] = id[t]; }
  __syncthreads();

  int ptr = 0;
  int* out = IDX + ((size_t)b * NPTS + i) * 16;
  for (int r = 0; r < 16; ++r) {
    float cand = lval[lane * 17 + ptr];
    int cid = lid[lane * 17 + ptr];
    int sl = lane;
#pragma unroll
    for (int off = 16; off > 0; off >>= 1) {
      float ov = __shfl_xor(cand, off, 32);
      int oid = __shfl_xor(cid, off, 32);
      int osl = __shfl_xor(sl, off, 32);
      if (ov < cand || (ov == cand && osl < sl)) { cand = ov; cid = oid; sl = osl; }
    }
    if (lane == sl && ptr < 16) ptr++;
    if (lane == 0) out[r] = cid;
  }
}

// --------------------------------------------- f16 WMMA fragment loads (LDS)
// A: weight rows are row-major; each VGPR's (k,k+1) pair is contiguous -> b32.
__device__ __forceinline__ v16h load_afrag(const _Float16* __restrict__ Wrow,
                                           int kb, int h) {
  union { v16h v; unsigned int u[8]; } fr;
#pragma unroll
  for (int t = 0; t < 8; ++t) {
    int ka = kb + ((t < 4) ? 2 * t : 16 + 2 * (t - 4)) + 8 * h;
    fr.u[t] = *(const unsigned int*)(Wrow + ka);
  }
  return fr.v;
}
// B: pair-interleaved activation matrix -> one aligned b32 per fragment VGPR.
template <int LDC>
__device__ __forceinline__ v16h load_bfrag(const _Float16* __restrict__ Bm,
                                           int kb, int col, int h) {
  union { v16h v; unsigned int u[8]; } fr;
#pragma unroll
  for (int t = 0; t < 8; ++t) {
    int kv = kb + ((t < 4) ? 2 * t : 16 + 2 * (t - 4)) + 8 * h;      // always even
    fr.u[t] = *(const unsigned int*)(Bm + (size_t)(kv >> 1) * (2 * LDC) + 2 * col);
  }
  return fr.v;
}

// ----------------------------------------------------- comp (1x1) conv, f16 WMMA
// block 128 thr (4 waves), grid (NPTS/64, NB). LDS: W[48][Kp], X interleaved [Kp][64].
template <int Cin>
__global__ void conv1x1_wmma_kernel(const float* __restrict__ X,
                                    const float* __restrict__ W,
                                    const float* __restrict__ Bias,
                                    float* __restrict__ Y) {
  constexpr int Kp = rup32c(Cin);
  constexpr int Cout = 48;
  extern __shared__ char smem[];
  _Float16* Ws = (_Float16*)smem;              // [48][Kp] row-major
  _Float16* Xs = Ws + (size_t)48 * Kp;         // [Kp][64] pair-interleaved
  const int b = blockIdx.y, n0 = blockIdx.x * 64, tid = threadIdx.x;

  for (int e = tid; e < 48 * Kp; e += 128) {
    int mm = e / Kp, k = e - mm * Kp;
    Ws[e] = (mm < Cout && k < Cin) ? (_Float16)W[mm * Cin + k] : (_Float16)0.f;
  }
  for (int e = tid; e < Kp * 64; e += 128) {
    int k = e >> 6, c = e & 63;
    Xs[ipos<64>(k, c)] =
        (k < Cin) ? (_Float16)X[((size_t)b * Cin + k) * NPTS + n0 + c]
                  : (_Float16)0.f;
  }
  __syncthreads();

  const int wv = tid >> 5, lane = tid & 31, m = lane & 15, h = lane >> 4;
  constexpr int Mt = (Cout + 15) >> 4;
  for (int job = wv; job < Mt * 4; job += 4) {
    const int mt = job >> 2, nt = job & 3;
    v8f acc;
#pragma unroll
    for (int r = 0; r < 8; ++r) {
      int oc = mt * 16 + r + 8 * h;
      acc[r] = (oc < Cout) ? Bias[oc] : 0.f;
    }
#pragma unroll
    for (int kb = 0; kb < Kp; kb += 32) {
      v16h a = load_afrag(Ws + (size_t)(mt * 16 + m) * Kp, kb, h);
      v16h bf = load_bfrag<64>(Xs, kb, nt * 16 + m, h);
      acc = __builtin_amdgcn_wmma_f32_16x16x32_f16(false, a, false, bf,
                                                   (short)0, acc, false, false);
    }
#pragma unroll
    for (int r = 0; r < 8; ++r) {
      int oc = mt * 16 + r + 8 * h;
      if (oc < Cout) {
        float v = acc[r];
        v = v > 0.f ? v : 0.f;                 // comp convs use relu
        Y[((size_t)b * Cout + oc) * NPTS + n0 + nt * 16 + m] = v;
      }
    }
  }
}

// ------------------------------------------------- fused denseconv, f16 WMMA
// cols: col = point*16 + neighbor (256 cols per block of 16 points)
template <int Kp>
__device__ __forceinline__ void dc_gemm(const _Float16* __restrict__ Ws,
                                        const _Float16* __restrict__ Bm,
                                        const float* __restrict__ bias,
                                        _Float16* __restrict__ Dst, bool relu,
                                        int tid) {
  const int wv = tid >> 5, lane = tid & 31, m = lane & 15, h = lane >> 4;
  for (int job = wv; job < 32; job += 4) {      // 2 mtiles * 16 ntiles
    const int mt = job >> 4, nt = job & 15;
    v8f acc;
#pragma unroll
    for (int r = 0; r < 8; ++r) {
      int oc = mt * 16 + r + 8 * h;
      acc[r] = (oc < Gc) ? bias[oc] : 0.f;
    }
#pragma unroll
    for (int kb = 0; kb < Kp; kb += 32) {
      v16h a = load_afrag(Ws + (size_t)(mt * 16 + m) * Kp, kb, h);
      v16h bf = load_bfrag<256>(Bm, kb, nt * 16 + m, h);
      acc = __builtin_amdgcn_wmma_f32_16x16x32_f16(false, a, false, bf,
                                                   (short)0, acc, false, false);
    }
#pragma unroll
    for (int r = 0; r < 8; ++r) {
      int oc = mt * 16 + r + 8 * h;
      if (oc < Gc) {
        float v = acc[r];
        if (relu) v = v > 0.f ? v : 0.f;
        Dst[ipos<256>(oc, nt * 16 + m)] = (_Float16)v;
      }
    }
  }
}

template <int Cin>
__global__ void denseconv_wmma_kernel(
    const float* __restrict__ X, const int* __restrict__ IDX,
    const float* __restrict__ W1, const float* __restrict__ B1,
    const float* __restrict__ W2, const float* __restrict__ B2,
    const float* __restrict__ W3, const float* __restrict__ B3,
    float* __restrict__ Y, int Ctot) {
  constexpr int K1 = 2 * Cin;          // even
  constexpr int K2 = Gc + Cin;
  constexpr int K3 = 2 * Gc + Cin;     // even
  constexpr int K1p = rup32c(K1);
  constexpr int K2p = rup32c(K2);
  constexpr int K3p = rup32c(K3);
  constexpr int R = (K3p > Gc + K2p) ? K3p : Gc + K2p;   // even

  extern __shared__ char smem[];
  _Float16* Ws1 = (_Float16*)smem;
  _Float16* Ws2 = Ws1 + (size_t)32 * K1p;
  _Float16* Ws3 = Ws2 + (size_t)32 * K2p;
  _Float16* YED = Ws3 + (size_t)32 * K3p;        // [K1p][256] interleaved
  _Float16* ACT = YED + (size_t)K1p * 256;       // [R][256] interleaved: h2|h1|central|pad

  const int b = blockIdx.y, n0 = blockIdx.x * 16, tid = threadIdx.x;

  for (int e = tid; e < 32 * K1p; e += 128) {
    int mm = e / K1p, k = e - mm * K1p;
    Ws1[e] = (mm < Gc && k < K1) ? (_Float16)W1[mm * K1 + k] : (_Float16)0.f;
  }
  for (int e = tid; e < 32 * K2p; e += 128) {
    int mm = e / K2p, k = e - mm * K2p;
    Ws2[e] = (mm < Gc && k < K2) ? (_Float16)W2[mm * K2 + k] : (_Float16)0.f;
  }
  for (int e = tid; e < 32 * K3p; e += 128) {
    int mm = e / K3p, k = e - mm * K3p;
    Ws3[e] = (mm < Gc && k < K3) ? (_Float16)W3[mm * K3 + k] : (_Float16)0.f;
  }
  // pad regions start at even rows -> same flat ranges in interleaved layout
  for (int e = tid; e < (K1p - K1) * 256; e += 128) YED[K1 * 256 + e] = (_Float16)0.f;
  for (int e = tid; e < (R - K3) * 256; e += 128)   ACT[K3 * 256 + e] = (_Float16)0.f;

  const float* Xb = X + (size_t)b * Cin * NPTS;
  const int* idxp = IDX + ((size_t)b * NPTS + n0) * 16;
  for (int e = tid; e < Cin * 256; e += 128) {
    int c = e >> 8, col = e & 255;
    int p = col >> 4, j = col & 15;
    float cv = Xb[(size_t)c * NPTS + n0 + p];
    int nb = idxp[p * 16 + j];
    float nv = Xb[(size_t)c * NPTS + nb];
    _Float16 ch = (_Float16)cv;
    YED[ipos<256>(c, col)] = ch;
    YED[ipos<256>(Cin + c, col)] = (_Float16)(nv - cv);
    ACT[ipos<256>(2 * Gc + c, col)] = ch;
  }
  __syncthreads();

  dc_gemm<K1p>(Ws1, YED, B1, ACT + (size_t)Gc * 256, true, tid);   // h1
  __syncthreads();
  dc_gemm<K2p>(Ws2, ACT + (size_t)Gc * 256, B2, ACT, true, tid);   // h2
  __syncthreads();
  dc_gemm<K3p>(Ws3, ACT, B3, YED, false, tid);                     // h3 -> YED
  __syncthreads();

  // epilogue: max over the 16 neighbor columns of each point
  constexpr int CoutTot = 3 * Gc + Cin;
  for (int e = tid; e < CoutTot * 16; e += 128) {
    int ch = e >> 4, p = e & 15;
    float v;
    if (ch >= 3 * Gc) {
      v = Xb[(size_t)(ch - 3 * Gc) * NPTS + n0 + p];   // central passthrough
    } else {
      const _Float16* r0 = (ch < Gc) ? (YED + ipos<256>(ch, p * 16))
                                     : (ACT + ipos<256>(ch - Gc, p * 16));
      v = (float)r0[0];
#pragma unroll
      for (int j = 1; j < 16; ++j) {
        float t = (float)r0[2 * j];              // interleaved col stride = 2
        if (t > v) v = t;
      }
    }
    Y[((size_t)b * Ctot + ch) * NPTS + n0 + p] = v;
  }
}

// ------------------------------------------------------------------ host side
template <int Cin>
static size_t dc_smem_bytes() {
  constexpr int K1p = rup32c(2 * Cin);
  constexpr int K2p = rup32c(Gc + Cin);
  constexpr int K3p = rup32c(2 * Gc + Cin);
  constexpr int R = (K3p > Gc + K2p) ? K3p : Gc + K2p;
  return ((size_t)32 * (K1p + K2p + K3p) + (size_t)K1p * 256 +
          (size_t)R * 256) * 2;
}

extern "C" void kernel_launch(void* const* d_in, const int* in_sizes, int n_in,
                              void* d_out, int out_size, void* d_ws, size_t ws_size,
                              hipStream_t stream) {
  (void)in_sizes; (void)n_in; (void)out_size; (void)ws_size;
  const float* x    = (const float*)d_in[0];
  const float* c1w  = (const float*)d_in[1];
  const float* c1b  = (const float*)d_in[2];
  const float* dc1p[6] = {(const float*)d_in[3],(const float*)d_in[4],(const float*)d_in[5],
                          (const float*)d_in[6],(const float*)d_in[7],(const float*)d_in[8]};
  const float* c2w  = (const float*)d_in[9];
  const float* c2b  = (const float*)d_in[10];
  const float* dc2p[6] = {(const float*)d_in[11],(const float*)d_in[12],(const float*)d_in[13],
                          (const float*)d_in[14],(const float*)d_in[15],(const float*)d_in[16]};
  const float* c3w  = (const float*)d_in[17];
  const float* c3b  = (const float*)d_in[18];
  const float* dc3p[6] = {(const float*)d_in[19],(const float*)d_in[20],(const float*)d_in[21],
                          (const float*)d_in[22],(const float*)d_in[23],(const float*)d_in[24]};
  const float* c4w  = (const float*)d_in[25];
  const float* c4b  = (const float*)d_in[26];
  const float* dc4p[6] = {(const float*)d_in[27],(const float*)d_in[28],(const float*)d_in[29],
                          (const float*)d_in[30],(const float*)d_in[31],(const float*)d_in[32]};

  float* ws = (float*)d_ws;
  float* l0 = ws; ws += (size_t)NB * 24  * NPTS;
  float* l1 = ws; ws += (size_t)NB * 120 * NPTS;
  float* c2 = ws; ws += (size_t)NB * 48  * NPTS;
  float* l2 = ws; ws += (size_t)NB * 240 * NPTS;
  float* c3 = ws; ws += (size_t)NB * 48  * NPTS;
  float* l3 = ws; ws += (size_t)NB * 360 * NPTS;
  float* c4 = ws; ws += (size_t)NB * 48  * NPTS;
  float* P  = ws; ws += (size_t)NPTS * NPTS;
  float* SQ = ws; ws += (size_t)NB * NPTS;
  int*  IDX = (int*)ws;

  float* out = (float*)d_out;

  auto knn24 = [&](const float* F) {
    sqnorm_kernel<24><<<(NB * NPTS + 255) / 256, 256, 0, stream>>>(F, SQ);
    for (int b = 0; b < NB; ++b) {
      knn_dist_kernel<24><<<dim3(NPTS / 16, NPTS / 128), 32, 0, stream>>>(F, b, P);
      knn_topk_kernel<<<NPTS, 32, 0, stream>>>(P, SQ, b, IDX);
    }
  };
  auto knn48 = [&](const float* F) {
    sqnorm_kernel<48><<<(NB * NPTS + 255) / 256, 256, 0, stream>>>(F, SQ);
    for (int b = 0; b < NB; ++b) {
      knn_dist_kernel<48><<<dim3(NPTS / 16, NPTS / 128), 32, 0, stream>>>(F, b, P);
      knn_topk_kernel<<<NPTS, 32, 0, stream>>>(P, SQ, b, IDX);
    }
  };
  auto cpy = [&](const float* S, int Cs, float* D, int Ctot, int off) {
    copych_kernel<<<((size_t)NB * Cs * NPTS + 255) / 256, 256, 0, stream>>>(
        S, Cs, D, Ctot, off);
  };

  // stage 0
  conv1_scalar_kernel<<<(NB * 24 * NPTS + 255) / 256, 256, 0, stream>>>(x, c1w, c1b, l0);

  // stage 1: dc1 on l0 (24ch) -> l1[0..95], concat l0 -> l1[96..119]
  knn24(l0);
  denseconv_wmma_kernel<24><<<dim3(NPTS / 16, NB), 128, dc_smem_bytes<24>(), stream>>>(
      l0, IDX, dc1p[0], dc1p[1], dc1p[2], dc1p[3], dc1p[4], dc1p[5], l1, 120);
  cpy(l0, 24, l1, 120, 96);

  // stage 2
  conv1x1_wmma_kernel<120><<<dim3(NPTS / 64, NB), 128,
      (size_t)(48 * rup32c(120) + rup32c(120) * 64) * 2, stream>>>(l1, c2w, c2b, c2);
  knn48(c2);
  denseconv_wmma_kernel<48><<<dim3(NPTS / 16, NB), 128, dc_smem_bytes<48>(), stream>>>(
      c2, IDX, dc2p[0], dc2p[1], dc2p[2], dc2p[3], dc2p[4], dc2p[5], l2, 240);
  cpy(l1, 120, l2, 240, 120);

  // stage 3
  conv1x1_wmma_kernel<240><<<dim3(NPTS / 64, NB), 128,
      (size_t)(48 * rup32c(240) + rup32c(240) * 64) * 2, stream>>>(l2, c3w, c3b, c3);
  knn48(c3);
  denseconv_wmma_kernel<48><<<dim3(NPTS / 16, NB), 128, dc_smem_bytes<48>(), stream>>>(
      c3, IDX, dc3p[0], dc3p[1], dc3p[2], dc3p[3], dc3p[4], dc3p[5], l3, 360);
  cpy(l2, 240, l3, 360, 120);

  // stage 4
  conv1x1_wmma_kernel<360><<<dim3(NPTS / 64, NB), 128,
      (size_t)(48 * rup32c(360) + rup32c(360) * 64) * 2, stream>>>(l3, c4w, c4b, c4);
  knn48(c4);
  denseconv_wmma_kernel<48><<<dim3(NPTS / 16, NB), 128, dc_smem_bytes<48>(), stream>>>(
      c4, IDX, dc4p[0], dc4p[1], dc4p[2], dc4p[3], dc4p[4], dc4p[5], out, 480);
  cpy(l3, 360, out, 480, 120);
}
